// Drug_fusion_19404662243407
// MI455X (gfx1250) — compile-verified, matrix operand order checked by reading, not run on hardware
//
#include <hip/hip_runtime.h>

// ---------------------------------------------------------------------------
// MI455X (gfx1250) implementation: wave32 + v_wmma_f32_16x16x32_f16 pipeline
//   K0: l2-normalize -> f16 operands + inverse norms
//   K1: aff = 0.5 * d1n @ d2n^T            (WMMA, 128x128 tiles)
//   K2: online softmax stats per row / per column of aff
//   K3: drug2_layer = softmax_row(aff) @ d2n   (WMMA, exp built on the fly)
//   K4: drug1_layer = softmax_col(aff)^T @ d1n (WMMA, transposed exp tile)
//   K5: head: fusion @ W^T + b -> relu -> softmax(64) ; avgpool4(fusion) * w
// ---------------------------------------------------------------------------

typedef __attribute__((ext_vector_type(16))) _Float16 v16h;
typedef __attribute__((ext_vector_type(8)))  _Float16 v8h;
typedef __attribute__((ext_vector_type(4)))  _Float16 v4h;
typedef __attribute__((ext_vector_type(8)))  float    v8f;

#define NSEQ 1024
#define DDIM 128
#define BATCH 32

__device__ __forceinline__ v8f wmma_f16(v16h a, v16h b, v8f c) {
  // D = A(16x32 f16) x B(32x16 f16) + C(16x16 f32)
  return __builtin_amdgcn_wmma_f32_16x16x32_f16(false, a, false, b, (short)0, c,
                                                false, false);
}

// Build one A/B fragment from an LDS tile laid out as [idx][K] (halfs).
// ISA 7.12.2 16-bit A layout: lane<16 -> K 0..7 (v0..3) and 16..23 (v4..7);
// lane>=16 -> K 8..15 and 24..31. stride_h and kk must keep 16B alignment.
__device__ __forceinline__ v16h load_frag(const _Float16* base, int stride_h,
                                          int idx0, int kk, int lane) {
  const int l  = lane & 15;
  const int hi = (lane >> 4) & 1;
  const _Float16* p = base + (size_t)(idx0 + l) * stride_h + kk + hi * 8;
  v8h lo = *(const v8h*)(p);
  v8h hh = *(const v8h*)(p + 16);
  v16h r;
#pragma unroll
  for (int i = 0; i < 8; ++i) { r[i] = lo[i]; r[8 + i] = hh[i]; }
  return r;
}

// ---------------------------------------------------------------------------
// K0: per-row l2 normalize (one wave32 per 128-float row)
// ---------------------------------------------------------------------------
__global__ __launch_bounds__(256) void normalize_kernel(
    const float* __restrict__ drug1, const float* __restrict__ drug2,
    _Float16* __restrict__ d1h, _Float16* __restrict__ d2h,
    float* __restrict__ inv1, float* __restrict__ inv2) {
  const int row  = blockIdx.x * 8 + (threadIdx.x >> 5);
  const int lane = threadIdx.x & 31;
  const int total = BATCH * NSEQ;
  const float* src; _Float16* dst; float* invp; int r;
  if (row < total) { src = drug1; dst = d1h; invp = inv1; r = row; }
  else             { src = drug2; dst = d2h; invp = inv2; r = row - total; }

  const float4 v = ((const float4*)(src + (size_t)r * DDIM))[lane];
  float ss = v.x * v.x + v.y * v.y + v.z * v.z + v.w * v.w;
#pragma unroll
  for (int k = 16; k >= 1; k >>= 1) ss += __shfl_xor(ss, k, 32);
  const float inv = 1.0f / fmaxf(sqrtf(ss), 1e-12f);
  if (lane == 0) invp[r] = inv;
  v4h h;
  h[0] = (_Float16)(v.x * inv); h[1] = (_Float16)(v.y * inv);
  h[2] = (_Float16)(v.z * inv); h[3] = (_Float16)(v.w * inv);
  ((v4h*)(dst + (size_t)r * DDIM))[lane] = h;
}

// ---------------------------------------------------------------------------
// K1: aff[b] = 0.5 * d1n[b] @ d2n[b]^T ; 128x128 output tile per block
// ---------------------------------------------------------------------------
__global__ __launch_bounds__(256) void aff_gemm_kernel(
    const _Float16* __restrict__ d1h, const _Float16* __restrict__ d2h,
    float* __restrict__ aff) {
  __shared__ alignas(16) _Float16 Ah[128][136];
  __shared__ alignas(16) _Float16 Bh[128][136];
  const int tid = threadIdx.x, lane = tid & 31, w = tid >> 5;
  const int n0 = blockIdx.x * 128, m0 = blockIdx.y * 128, b = blockIdx.z;

  const uint4* ga = (const uint4*)(d1h + ((size_t)b * NSEQ + n0) * DDIM);
  const uint4* gb = (const uint4*)(d2h + ((size_t)b * NSEQ + m0) * DDIM);
  for (int e = tid; e < 128 * 16; e += 256) {
    const int r = e >> 4, q = e & 15;
    *(uint4*)&Ah[r][q * 8] = ga[r * 16 + q];
    *(uint4*)&Bh[r][q * 8] = gb[r * 16 + q];
  }
  __syncthreads();

  const int rg = (w >> 1) * 32;   // 32-row band
  const int cg = (w & 1) * 64;    // 64-col band
  v8f acc[2][4];
#pragma unroll
  for (int t = 0; t < 2; ++t)
#pragma unroll
    for (int j = 0; j < 4; ++j) acc[t][j] = (v8f){0, 0, 0, 0, 0, 0, 0, 0};

#pragma unroll
  for (int kk = 0; kk < DDIM; kk += 32) {
    const v16h a0 = load_frag(&Ah[0][0], 136, rg, kk, lane);
    const v16h a1 = load_frag(&Ah[0][0], 136, rg + 16, kk, lane);
#pragma unroll
    for (int j = 0; j < 4; ++j) {
      const v16h bf = load_frag(&Bh[0][0], 136, cg + j * 16, kk, lane);
      acc[0][j] = wmma_f16(a0, bf, acc[0][j]);
      acc[1][j] = wmma_f16(a1, bf, acc[1][j]);
    }
  }

  float* outp = aff + ((size_t)b * NSEQ + n0) * NSEQ + m0;
  const int col = lane & 15, rb = (lane >> 4) * 8;
#pragma unroll
  for (int t = 0; t < 2; ++t)
#pragma unroll
    for (int j = 0; j < 4; ++j)
#pragma unroll
      for (int i = 0; i < 8; ++i)
        outp[(size_t)(rg + t * 16 + rb + i) * NSEQ + cg + j * 16 + col] =
            acc[t][j][i] * 0.5f;
}

// ---------------------------------------------------------------------------
// K2a: online (max,sum) per row of aff
// ---------------------------------------------------------------------------
__global__ __launch_bounds__(256) void row_stats_kernel(
    const float* __restrict__ aff, float* __restrict__ rmax,
    float* __restrict__ rsum) {
  __shared__ float red_m[8], red_s[8];
  const int n = blockIdx.x, b = blockIdx.y;
  const int tid = threadIdx.x, lane = tid & 31, w = tid >> 5;
  const float* row = aff + ((size_t)b * NSEQ + n) * NSEQ;
  float m = -3.4e38f, s = 0.0f;
  for (int i = tid; i < NSEQ; i += 256) {
    const float x = row[i];
    const float nm = fmaxf(m, x);
    s = s * __expf(m - nm) + __expf(x - nm);
    m = nm;
  }
#pragma unroll
  for (int k = 16; k >= 1; k >>= 1) {
    const float m2 = __shfl_xor(m, k, 32), s2 = __shfl_xor(s, k, 32);
    const float nm = fmaxf(m, m2);
    s = s * __expf(m - nm) + s2 * __expf(m2 - nm);
    m = nm;
  }
  if (lane == 0) { red_m[w] = m; red_s[w] = s; }
  __syncthreads();
  if (tid == 0) {
    m = red_m[0]; s = red_s[0];
    for (int j = 1; j < 8; ++j) {
      const float nm = fmaxf(m, red_m[j]);
      s = s * __expf(m - nm) + red_s[j] * __expf(red_m[j] - nm);
      m = nm;
    }
    rmax[(size_t)b * NSEQ + n] = m;
    rsum[(size_t)b * NSEQ + n] = s;
  }
}

// ---------------------------------------------------------------------------
// K2b: online (max,sum) per column of aff (64 columns per block, coalesced)
// ---------------------------------------------------------------------------
__global__ __launch_bounds__(256) void col_stats_kernel(
    const float* __restrict__ aff, float* __restrict__ cmax,
    float* __restrict__ csum) {
  __shared__ float red_m[4][64], red_s[4][64];
  const int b = blockIdx.y, m0 = blockIdx.x * 64;
  const int tid = threadIdx.x, c = tid & 63, rg = tid >> 6;
  const float* base = aff + (size_t)b * NSEQ * NSEQ + m0 + c;
  float m = -3.4e38f, s = 0.0f;
  for (int n = rg; n < NSEQ; n += 4) {
    const float x = base[(size_t)n * NSEQ];
    const float nm = fmaxf(m, x);
    s = s * __expf(m - nm) + __expf(x - nm);
    m = nm;
  }
  red_m[rg][c] = m; red_s[rg][c] = s;
  __syncthreads();
  if (rg == 0) {
    for (int j = 1; j < 4; ++j) {
      const float nm = fmaxf(m, red_m[j][c]);
      s = s * __expf(m - nm) + red_s[j][c] * __expf(red_m[j][c] - nm);
      m = nm;
    }
    cmax[(size_t)b * NSEQ + m0 + c] = m;
    csum[(size_t)b * NSEQ + m0 + c] = s;
  }
}

// ---------------------------------------------------------------------------
// K3/K4: out[b, r, :] = (1/sum[r]) * sum_k exp(aff(..)-max[r]) * V[b, k, :]
// transA=0: rows are n, aff read row-major    (drug2_layer = s1 @ d2n)
// transA=1: rows are m, aff read transposed   (drug1_layer = s2^T @ d1n)
// ---------------------------------------------------------------------------
__global__ __launch_bounds__(256) void pv_gemm_kernel(
    const float* __restrict__ aff, const float* __restrict__ smax,
    const float* __restrict__ ssum, const _Float16* __restrict__ vh,
    float* __restrict__ outlayer, int transA) {
  __shared__ alignas(16) _Float16 Eh[128][40];   // exp tile   [row][k]
  __shared__ alignas(16) _Float16 Vt[128][40];   // V^T tile   [d][k]
  const int tid = threadIdx.x, lane = tid & 31, w = tid >> 5;
  const int r0 = blockIdx.x * 128, b = blockIdx.y;
  const float* affb = aff + (size_t)b * NSEQ * NSEQ;
  const float* smx = smax + (size_t)b * NSEQ;

  v8f acc[8];
#pragma unroll
  for (int j = 0; j < 8; ++j) acc[j] = (v8f){0, 0, 0, 0, 0, 0, 0, 0};

  for (int kk = 0; kk < NSEQ; kk += 32) {
    if (!transA) {
      for (int e = tid; e < 128 * 32; e += 256) {
        const int r = e >> 5, k = e & 31;
        const float x = affb[(size_t)(r0 + r) * NSEQ + kk + k];
        Eh[r][k] = (_Float16)__expf(x - smx[r0 + r]);
      }
    } else {
      for (int e = tid; e < 32 * 128; e += 256) {
        const int k = e >> 7, mm = e & 127;
        const float x = affb[(size_t)(kk + k) * NSEQ + r0 + mm];
        Eh[mm][k] = (_Float16)__expf(x - smx[r0 + mm]);
      }
    }
    for (int e = tid; e < 32 * 128; e += 256) {
      const int k = e >> 7, d = e & 127;
      Vt[d][k] = vh[((size_t)b * NSEQ + kk + k) * DDIM + d];
    }
    __syncthreads();
    const v16h a = load_frag(&Eh[0][0], 40, w * 16, 0, lane);
#pragma unroll
    for (int j = 0; j < 8; ++j) {
      const v16h bf = load_frag(&Vt[0][0], 40, j * 16, 0, lane);
      acc[j] = wmma_f16(a, bf, acc[j]);
    }
    __syncthreads();
  }

  const float* ssm = ssum + (size_t)b * NSEQ;
  const int col = lane & 15, rb = (lane >> 4) * 8;
#pragma unroll
  for (int j = 0; j < 8; ++j)
#pragma unroll
    for (int i = 0; i < 8; ++i) {
      const int rr = r0 + w * 16 + rb + i;
      outlayer[((size_t)b * NSEQ + rr) * DDIM + j * 16 + col] =
          acc[j][i] * (1.0f / ssm[rr]);
    }
}

// ---------------------------------------------------------------------------
// K5: head. 128 fusion rows per block. WMMA 16x64 = fusion(16x256) @ W^T.
// ---------------------------------------------------------------------------
__global__ __launch_bounds__(256) void head_kernel(
    const float* __restrict__ drug1, const float* __restrict__ drug2,
    const float* __restrict__ inv1, const float* __restrict__ inv2,
    const float* __restrict__ l1, const float* __restrict__ l2,
    const float* __restrict__ fw, const float* __restrict__ fb,
    float* __restrict__ out) {
  __shared__ alignas(16) _Float16 Fh[128][264];  // fusion tile [row][k]
  __shared__ alignas(16) _Float16 Wt[64][264];   // W as [o][k]
  __shared__ float Wo[128][68];                  // relu(linear) results
  __shared__ float bs[64];
  const int tid = threadIdx.x, lane = tid & 31, w = tid >> 5;
  const int r0 = blockIdx.x * 128, b = blockIdx.y;

  // entire block is in the same half of fusion (r0 multiple of 128)
  const int half = (r0 >= NSEQ);
  const float* dsrc = half ? drug2 : drug1;
  const float* isrc = half ? inv2 : inv1;
  const float* lsrc = half ? l1 : l2;
  const int rbase = half ? (r0 - NSEQ) : r0;

  for (int e = tid; e < 64 * 256; e += 256) Wt[e >> 8][e & 255] = (_Float16)fw[e];
  if (tid < 64) bs[tid] = fb[tid];

  for (int e = tid; e < 128 * 256; e += 256) {
    const int r = e >> 8, k = e & 255;
    const size_t gr = (size_t)b * NSEQ + rbase + r;
    const float val = (k < 128) ? dsrc[gr * DDIM + k] * isrc[gr]
                                : lsrc[gr * DDIM + (k - 128)];
    Fh[r][k] = (_Float16)val;
  }
  __syncthreads();

  v8f acc[4];
#pragma unroll
  for (int j = 0; j < 4; ++j) acc[j] = (v8f){0, 0, 0, 0, 0, 0, 0, 0};
#pragma unroll
  for (int kk = 0; kk < 256; kk += 32) {
    const v16h a = load_frag(&Fh[0][0], 264, w * 16, kk, lane);
#pragma unroll
    for (int j = 0; j < 4; ++j) {
      const v16h bf = load_frag(&Wt[0][0], 264, j * 16, kk, lane);
      acc[j] = wmma_f16(a, bf, acc[j]);
    }
  }
  const int col0 = lane & 15, rb = (lane >> 4) * 8;
#pragma unroll
  for (int j = 0; j < 4; ++j)
#pragma unroll
    for (int i = 0; i < 8; ++i) {
      const int r = w * 16 + rb + i, o = j * 16 + col0;
      Wo[r][o] = fmaxf(acc[j][i] + bs[o], 0.0f);
    }
  __syncthreads();

  // softmax over 64 outputs + avgpool4(fusion) * w  (one wave per 16 rows)
  for (int rr = 0; rr < 16; ++rr) {
    const int r = w * 16 + rr;
    const size_t gr = (size_t)b * NSEQ + rbase + r;
    const float w1 = Wo[r][lane], w2 = Wo[r][lane + 32];
    float mx = fmaxf(w1, w2);
#pragma unroll
    for (int k = 16; k >= 1; k >>= 1) mx = fmaxf(mx, __shfl_xor(mx, k, 32));
    const float e1 = __expf(w1 - mx), e2 = __expf(w2 - mx);
    float s = e1 + e2;
#pragma unroll
    for (int k = 16; k >= 1; k >>= 1) s += __shfl_xor(s, k, 32);
    const float sinv = 1.0f / s;

    const float4 q1 = ((const float4*)(dsrc + gr * DDIM))[lane];
    const float iv = isrc[gr];
    const float p1 = (q1.x + q1.y + q1.z + q1.w) * 0.25f * iv;
    const float4 q2 = ((const float4*)(lsrc + gr * DDIM))[lane];
    const float p2 = (q2.x + q2.y + q2.z + q2.w) * 0.25f;

    float* op = out + ((size_t)b * (2 * NSEQ) + r0 + r) * 64;
    op[lane]      = p1 * e1 * sinv;
    op[lane + 32] = p2 * e2 * sinv;
  }
}

// ---------------------------------------------------------------------------
extern "C" void kernel_launch(void* const* d_in, const int* in_sizes, int n_in,
                              void* d_out, int out_size, void* d_ws,
                              size_t ws_size, hipStream_t stream) {
  (void)in_sizes; (void)n_in; (void)out_size; (void)ws_size;
  const float* drug1 = (const float*)d_in[0];
  const float* drug2 = (const float*)d_in[1];
  const float* fw    = (const float*)d_in[2];
  const float* fb    = (const float*)d_in[3];
  float* outp = (float*)d_out;

  const size_t BN = (size_t)BATCH * NSEQ;
  char* p = (char*)d_ws;
  auto carve = [&](size_t bytes) {
    char* q = p;
    p += (bytes + 255) & ~(size_t)255;
    return (void*)q;
  };
  _Float16* d1h = (_Float16*)carve(BN * DDIM * 2);
  _Float16* d2h = (_Float16*)carve(BN * DDIM * 2);
  float* inv1 = (float*)carve(BN * 4);
  float* inv2 = (float*)carve(BN * 4);
  float* rmax = (float*)carve(BN * 4);
  float* rsum = (float*)carve(BN * 4);
  float* cmax = (float*)carve(BN * 4);
  float* csum = (float*)carve(BN * 4);
  float* lay1 = (float*)carve(BN * DDIM * 4);  // drug1_layer [B,N2,D]
  float* lay2 = (float*)carve(BN * DDIM * 4);  // drug2_layer [B,N1,D]
  float* aff  = (float*)carve((size_t)BATCH * NSEQ * NSEQ * 4);

  normalize_kernel<<<dim3(2 * BN / 8), 256, 0, stream>>>(drug1, drug2, d1h, d2h,
                                                         inv1, inv2);
  aff_gemm_kernel<<<dim3(8, 8, BATCH), 256, 0, stream>>>(d1h, d2h, aff);
  row_stats_kernel<<<dim3(NSEQ, BATCH), 256, 0, stream>>>(aff, rmax, rsum);
  col_stats_kernel<<<dim3(16, BATCH), 256, 0, stream>>>(aff, cmax, csum);
  pv_gemm_kernel<<<dim3(8, BATCH), 256, 0, stream>>>(aff, rmax, rsum, d2h, lay2, 0);
  pv_gemm_kernel<<<dim3(8, BATCH), 256, 0, stream>>>(aff, cmax, csum, d1h, lay1, 1);
  head_kernel<<<dim3(16, BATCH), 256, 0, stream>>>(drug1, drug2, inv1, inv2,
                                                   lay1, lay2, fw, fb, outp);
}
